// TransformerDecoderLayer_26757646254622
// MI455X (gfx1250) — matrix operand order repaired and checked by reading, not compiled, not run on hardware
//
#include <hip/hip_runtime.h>
#include <hip/hip_bf16.h>
#include <cstdint>

// ---------------------------------------------------------------------------
// Types for CDNA5 WMMA (wave32): v_wmma_f32_16x16x32_f16
// ---------------------------------------------------------------------------
typedef _Float16 half_t;
typedef __attribute__((ext_vector_type(16))) _Float16 v16h;
typedef __attribute__((ext_vector_type(8)))  _Float16 v8h;
typedef __attribute__((ext_vector_type(8)))  float    v8f;

#define D_MODEL 256
#define NHEAD   8
#define HDIM    32
#define DFF     1024
#define BSZ     16
#define LQRY    300
#define LENV    13294
#define NROWS   (BSZ * LQRY)    // 4800
#define VROWS   (BSZ * LENV)    // 212704

__device__ __forceinline__ int imin(int a, int b) { return a < b ? a : b; }

// Combine two 16-byte LDS runs into one 16-half fragment (matches CDNA5
// A-matrix layout: elems 0..7 -> K = g*8 + e ; elems 8..15 -> K = 16 + g*8 + e
// and B read from the [n][k] tile, K contiguous per column, in the same K order)
__device__ __forceinline__ v16h make_frag(const half_t* lo, const half_t* hi) {
  v8h a = *(const v8h*)lo;
  v8h b = *(const v8h*)hi;
  v16h r;
#pragma unroll
  for (int i = 0; i < 8; ++i) { r[i] = a[i]; r[i + 8] = b[i]; }
  return r;
}

__device__ __forceinline__ v8f wmma_f16(v16h a, v16h b, v8f c) {
  return __builtin_amdgcn_wmma_f32_16x16x32_f16(false, a, false, b,
                                                (short)0, c, false, false);
}

// ---------------------------------------------------------------------------
// Generic GEMM: C[M,N] = act(A[M,K] @ B[K,N] + bias). fp32 in, f16 WMMA,
// fp32 accumulate, fp32 or f16 out.
// Block = 256 threads = 8 waves; block tile 128x64; each wave owns a 32x32
// patch = 2x2 WMMA tiles -> 4 WMMAs per 8 LDS fragment loads per k-step.
// ---------------------------------------------------------------------------
__global__ __launch_bounds__(256)
void gemm_wmma_kernel(const float* __restrict__ A, const float* __restrict__ B,
                      const float* __restrict__ bias,
                      float* __restrict__ Cf, half_t* __restrict__ Ch,
                      int M, int N, int K, int relu)
{
  __shared__ __align__(16) half_t As[128][40];   // row-major, K contiguous
  __shared__ __align__(16) half_t BsT[64][40];   // [n][k], K contiguous

  const int tid  = threadIdx.x;
  const int lane = tid & 31;
  const int wv   = tid >> 5;
  const int g    = lane >> 4;          // half-wave group
  const int lr   = lane & 15;
  const int wm   = (wv >> 1) * 32;     // 0,32,64,96 : wave's 32-row slab
  const int wn   = (wv & 1) * 32;      // 0,32       : wave's 32-col slab
  const int m0   = blockIdx.x * 128;
  const int n0   = blockIdx.y * 64;

  // staging indices: A 128x32 (16 elems/thread), B 32x64 (8 elems/thread)
  const int arow = tid >> 1;           // 0..127
  const int akc  = (tid & 1) * 16;     // 0,16
  const int bk   = tid >> 3;           // 0..31
  const int bn   = (tid & 7) * 8;      // 0..56

  v8f acc00 = {}, acc01 = {}, acc10 = {}, acc11 = {};

  for (int k0 = 0; k0 < K; k0 += 32) {
    // --- stage A tile (128x32) as f16 ---
    {
      const int gr = m0 + arow;
      if (gr < M) {
        const float* ap = A + (size_t)gr * K + k0 + akc;
        if (k0 + 32 < K) __builtin_prefetch(ap + 32, 0, 1);  // global_prefetch
#pragma unroll
        for (int i = 0; i < 16; ++i) As[arow][akc + i] = (half_t)ap[i];
      } else {
#pragma unroll
        for (int i = 0; i < 16; ++i) As[arow][akc + i] = (half_t)0.f;
      }
      // --- stage B tile (32x64), transposed so K is contiguous per column ---
      const float* bp = B + (size_t)(k0 + bk) * N + n0 + bn;
#pragma unroll
      for (int i = 0; i < 8; ++i) BsT[bn + i][bk] = (half_t)bp[i];
    }
    __syncthreads();

    const v16h a0 = make_frag(&As[wm + lr][g * 8],      &As[wm + lr][16 + g * 8]);
    const v16h a1 = make_frag(&As[wm + 16 + lr][g * 8], &As[wm + 16 + lr][16 + g * 8]);
    const v16h b0 = make_frag(&BsT[wn + lr][g * 8],      &BsT[wn + lr][16 + g * 8]);
    const v16h b1 = make_frag(&BsT[wn + 16 + lr][g * 8], &BsT[wn + 16 + lr][16 + g * 8]);

    acc00 = wmma_f16(a0, b0, acc00);
    acc01 = wmma_f16(a0, b1, acc01);
    acc10 = wmma_f16(a1, b0, acc10);
    acc11 = wmma_f16(a1, b1, acc11);
    __syncthreads();
  }

  // D layout: lane -> col N = lr, elem v -> row M = v + g*8
  const int col0 = n0 + wn + lr;
  const int col1 = col0 + 16;
  const float bv0 = bias ? bias[col0] : 0.f;
  const float bv1 = bias ? bias[col1] : 0.f;
#pragma unroll
  for (int v = 0; v < 8; ++v) {
    const int row0 = m0 + wm + g * 8 + v;
    const int row1 = row0 + 16;
    if (row0 < M) {
      float o00 = acc00[v] + bv0, o01 = acc01[v] + bv1;
      if (relu) { o00 = fmaxf(o00, 0.f); o01 = fmaxf(o01, 0.f); }
      if (Cf) { Cf[(size_t)row0 * N + col0] = o00; Cf[(size_t)row0 * N + col1] = o01; }
      else    { Ch[(size_t)row0 * N + col0] = (half_t)o00; Ch[(size_t)row0 * N + col1] = (half_t)o01; }
    }
    if (row1 < M) {
      float o10 = acc10[v] + bv0, o11 = acc11[v] + bv1;
      if (relu) { o10 = fmaxf(o10, 0.f); o11 = fmaxf(o11, 0.f); }
      if (Cf) { Cf[(size_t)row1 * N + col0] = o10; Cf[(size_t)row1 * N + col1] = o11; }
      else    { Ch[(size_t)row1 * N + col0] = (half_t)o10; Ch[(size_t)row1 * N + col1] = (half_t)o11; }
    }
  }
}

// ---------------------------------------------------------------------------
// Fused self-attention per (b, h, 64 q-rows). 4 independent waves, each owns
// 16 q rows. Scores: q(16x32) @ K^T via 20 WMMAs (HD=32 == one k-step).
// Softmax in LDS (f16), then probs(16x320) @ V(320x32) via 20 WMMAs.
// ---------------------------------------------------------------------------
__global__ __launch_bounds__(128)
void attn_wmma_kernel(const float* __restrict__ qb, const float* __restrict__ kb,
                      const float* __restrict__ vb, float* __restrict__ ob)
{
  __shared__ __align__(16) half_t S[4][16][328];  // per-wave 16x320 scores/probs

  const int tid  = threadIdx.x;
  const int lane = tid & 31;
  const int wv   = tid >> 5;
  const int g    = lane >> 4;
  const int lr   = lane & 15;

  int bid = blockIdx.x;
  const int qblk = bid % 5; bid /= 5;
  const int h = bid % NHEAD;
  const int b = bid / NHEAD;
  const int q0 = qblk * 64 + wv * 16;
  const float scale = 0.1767766953f;  // 1/sqrt(32)

  // A fragment: this wave's 16 q rows (scaled), clamp OOB rows (stores guarded)
  const int qr = imin(q0 + lr, LQRY - 1);
  const float* qp = qb + ((size_t)(b * LQRY + qr)) * D_MODEL + h * HDIM;
  v16h aq;
#pragma unroll
  for (int e = 0; e < 8; ++e) {
    aq[e]     = (half_t)(qp[g * 8 + e] * scale);
    aq[e + 8] = (half_t)(qp[16 + g * 8 + e] * scale);
  }

  // scores = q @ K^T, one WMMA per 16 keys
  for (int kt = 0; kt < 20; ++kt) {
    const int key = kt * 16 + lr;
    const int kc  = imin(key, LQRY - 1);
    const float* kp = kb + ((size_t)(b * LQRY + kc)) * D_MODEL + h * HDIM + g * 8;
    v16h bkf;
#pragma unroll
    for (int e = 0; e < 16; ++e) bkf[e] = (half_t)kp[(e & 8) * 2 + (e & 7)];  // K[key][g*8 + (e&7) + (e>=8 ? 16 : 0)]
    v8f s = {};
    s = wmma_f16(aq, bkf, s);
#pragma unroll
    for (int v = 0; v < 8; ++v) {
      const float sv = (key < LQRY) ? s[v] : -1e30f;  // mask padded keys
      S[wv][g * 8 + v][kt * 16 + lr] = (half_t)sv;
    }
  }
  __syncthreads();

  // softmax over each of the wave's 16 rows (lanes 0..15, one row each)
  if (lane < 16) {
    half_t* row = &S[wv][lane][0];
    float m = -1e30f;
    for (int k = 0; k < 320; ++k) m = fmaxf(m, (float)row[k]);
    float sum = 0.f;
    for (int k = 0; k < 320; ++k) {
      const float e = __expf((float)row[k] - m);
      sum += e; row[k] = (half_t)e;
    }
    const float inv = 1.f / sum;
    for (int k = 0; k < 320; ++k) row[k] = (half_t)((float)row[k] * inv);
  }
  __syncthreads();

  // out = probs @ V ; K = 320 keys -> 10 k-steps, 2 column tiles (HD = 32)
  v8f o0 = {}, o1 = {};
  for (int kt = 0; kt < 10; ++kt) {
    const half_t* pr = &S[wv][lr][kt * 32];
    v16h ap = make_frag(pr + g * 8, pr + 16 + g * 8);
    v16h bv0, bv1;
#pragma unroll
    for (int e = 0; e < 16; ++e) {
      const int ky = imin(kt * 32 + g * 8 + (e & 8) * 2 + (e & 7), LQRY - 1);  // prob==0 past 299
      const float* vp = vb + ((size_t)(b * LQRY + ky)) * D_MODEL + h * HDIM;
      bv0[e] = (half_t)vp[lr];
      bv1[e] = (half_t)vp[16 + lr];
    }
    o0 = wmma_f16(ap, bv0, o0);
    o1 = wmma_f16(ap, bv1, o1);
  }
#pragma unroll
  for (int v = 0; v < 8; ++v) {
    const int qrow = q0 + g * 8 + v;
    if (qrow < LQRY) {
      const size_t base = ((size_t)(b * LQRY + qrow)) * D_MODEL + h * HDIM;
      ob[base + lr]      = o0[v];
      ob[base + 16 + lr] = o1[v];
    }
  }
}

// ---------------------------------------------------------------------------
// Deformable sampling: one wave per (b,q,h); lane == channel d (HD==32==wave32)
// ---------------------------------------------------------------------------
__global__ __launch_bounds__(128)
void deform_sample_kernel(const float* __restrict__ so, const float* __restrict__ aw,
                          const float* __restrict__ rp, const half_t* __restrict__ val,
                          float* __restrict__ ca)
{
  const int item = blockIdx.x * 4 + (threadIdx.x >> 5);
  const int d    = threadIdx.x & 31;
  if (item >= BSZ * LQRY * NHEAD) return;
  const int h  = item & 7;
  const int bq = item >> 3;
  const int b  = bq / LQRY;

  const int Hs[4] = {100, 50, 25, 13};
  const int Wd[4] = {100, 50, 25, 13};
  const int st[4] = {0, 10000, 12500, 13125};

  // softmax of 16 attention weights (uniform across lanes -> scalar loads)
  const float* awp = aw + (size_t)bq * 128 + h * 16;
  float m = -1e30f;
#pragma unroll
  for (int i = 0; i < 16; ++i) m = fmaxf(m, awp[i]);
  float w[16]; float wsum = 0.f;
#pragma unroll
  for (int i = 0; i < 16; ++i) { w[i] = __expf(awp[i] - m); wsum += w[i]; }
  const float inv = 1.f / wsum;

  const float* sop = so + (size_t)bq * 256 + h * 32;
  float acc = 0.f;
#pragma unroll
  for (int l = 0; l < 4; ++l) {
    const int H = Hs[l], W = Wd[l];
    const float* rpp = rp + ((size_t)bq * 4 + l) * 4;
    const float cx = rpp[0], cy = rpp[1], cw = rpp[2], chh = rpp[3];
    const half_t* vb = val + ((size_t)(b * LENV + st[l])) * 256 + h * 32 + d;
#pragma unroll
    for (int p = 0; p < 4; ++p) {
      const float ox = sop[l * 8 + p * 2 + 0];
      const float oy = sop[l * 8 + p * 2 + 1];
      const float lx = cx + ox * 0.25f * cw * 0.5f;
      const float ly = cy + oy * 0.25f * chh * 0.5f;
      const float x = lx * W - 0.5f;
      const float y = ly * H - 0.5f;
      const float x0f = floorf(x), y0f = floorf(y);
      const float dx = x - x0f, dy = y - y0f;
      const int x0 = (int)x0f, y0 = (int)y0f;
      const float ww = w[l * 4 + p] * inv;
      const float cwt[4] = {(1 - dx) * (1 - dy), dx * (1 - dy), (1 - dx) * dy, dx * dy};
      const int xs[4] = {x0, x0 + 1, x0, x0 + 1};
      const int ys[4] = {y0, y0, y0 + 1, y0 + 1};
#pragma unroll
      for (int c = 0; c < 4; ++c) {
        if (xs[c] >= 0 && xs[c] < W && ys[c] >= 0 && ys[c] < H) {
          acc += ww * cwt[c] * (float)vb[(size_t)(ys[c] * W + xs[c]) * 256];
        }
      }
    }
  }
  ca[(size_t)bq * 256 + h * 32 + d] = acc;
}

// ---------------------------------------------------------------------------
// out = LayerNorm(x + res) * g + b   (one 256-thread block per row, D=256)
// ---------------------------------------------------------------------------
__global__ __launch_bounds__(256)
void add_ln_kernel(const float* __restrict__ x, const float* __restrict__ res,
                   const float* __restrict__ gam, const float* __restrict__ bet,
                   float* __restrict__ out)
{
  __shared__ float red[256];
  const int r = blockIdx.x, c = threadIdx.x;
  const size_t base = (size_t)r * 256;
  const float v = x[base + c] + res[base + c];
  red[c] = v; __syncthreads();
  for (int s = 128; s > 0; s >>= 1) { if (c < s) red[c] += red[c + s]; __syncthreads(); }
  const float mean = red[0] * (1.f / 256.f);
  __syncthreads();
  const float dv = v - mean;
  red[c] = dv * dv; __syncthreads();
  for (int s = 128; s > 0; s >>= 1) { if (c < s) red[c] += red[c + s]; __syncthreads(); }
  const float var = red[0] * (1.f / 256.f);
  out[base + c] = dv * rsqrtf(var + 1e-5f) * gam[c] + bet[c];
}

// ---------------------------------------------------------------------------
extern "C" void kernel_launch(void* const* d_in, const int* in_sizes, int n_in,
                              void* d_out, int out_size, void* d_ws, size_t ws_size,
                              hipStream_t stream)
{
  (void)in_sizes; (void)n_in; (void)out_size; (void)ws_size;

  const float* tgt  = (const float*)d_in[0];
  const float* rp   = (const float*)d_in[1];
  const float* mem  = (const float*)d_in[2];
  const float* Wq   = (const float*)d_in[3];
  const float* Wk   = (const float*)d_in[4];
  const float* Wv   = (const float*)d_in[5];
  const float* Wo   = (const float*)d_in[6];
  const float* Wvp  = (const float*)d_in[7];
  const float* Wso  = (const float*)d_in[8];
  const float* Waw  = (const float*)d_in[9];
  const float* Wop  = (const float*)d_in[10];
  const float* W1   = (const float*)d_in[11];
  const float* W2   = (const float*)d_in[12];
  const float* bq   = (const float*)d_in[13];
  const float* bk   = (const float*)d_in[14];
  const float* bv   = (const float*)d_in[15];
  const float* bo   = (const float*)d_in[16];
  const float* bvp  = (const float*)d_in[17];
  const float* bso  = (const float*)d_in[18];
  const float* baw  = (const float*)d_in[19];
  const float* bop  = (const float*)d_in[20];
  const float* bf1  = (const float*)d_in[21];
  const float* bf2  = (const float*)d_in[22];
  const float* ln1g = (const float*)d_in[23];
  const float* ln1b = (const float*)d_in[24];
  const float* ln2g = (const float*)d_in[25];
  const float* ln2b = (const float*)d_in[26];
  const float* ln3g = (const float*)d_in[27];
  const float* ln3b = (const float*)d_in[28];

  // workspace carve-out
  char* ws = (char*)d_ws;
  size_t off = 0;
  auto alloc = [&](size_t bytes) -> char* {
    char* p = ws + off;
    off += (bytes + 255) & ~(size_t)255;
    return p;
  };
  const size_t RB = (size_t)NROWS * 256 * sizeof(float);
  float*  qbuf = (float*)alloc(RB);
  float*  kbuf = (float*)alloc(RB);
  float*  vbuf = (float*)alloc(RB);
  float*  aout = (float*)alloc(RB);
  float*  sab  = (float*)alloc(RB);
  float*  t1   = (float*)alloc(RB);
  float*  sob  = (float*)alloc(RB);
  float*  awb  = (float*)alloc((size_t)NROWS * 128 * sizeof(float));
  float*  cab  = (float*)alloc(RB);
  float*  capb = (float*)alloc(RB);
  float*  t2   = (float*)alloc(RB);
  float*  ffh  = (float*)alloc((size_t)NROWS * DFF * sizeof(float));
  float*  ffo  = (float*)alloc(RB);
  half_t* valh = (half_t*)alloc((size_t)VROWS * 256 * sizeof(half_t));

  auto gemm = [&](const float* A, const float* B, const float* bias,
                  float* Cf, half_t* Ch, int M, int N, int K, int relu) {
    dim3 grid((M + 127) / 128, N / 64);
    gemm_wmma_kernel<<<grid, 256, 0, stream>>>(A, B, bias, Cf, Ch, M, N, K, relu);
  };

  // --- self attention ---
  gemm(tgt, Wq, bq, qbuf, nullptr, NROWS, 256, 256, 0);
  gemm(tgt, Wk, bk, kbuf, nullptr, NROWS, 256, 256, 0);
  gemm(tgt, Wv, bv, vbuf, nullptr, NROWS, 256, 256, 0);
  attn_wmma_kernel<<<BSZ * NHEAD * 5, 128, 0, stream>>>(qbuf, kbuf, vbuf, aout);
  gemm(aout, Wo, bo, sab, nullptr, NROWS, 256, 256, 0);
  add_ln_kernel<<<NROWS, 256, 0, stream>>>(sab, tgt, ln1g, ln1b, t1);

  // --- deformable cross attention ---
  gemm(mem, Wvp, bvp, nullptr, valh, VROWS, 256, 256, 0);  // dominant GEMM, f16 out
  gemm(t1, Wso, bso, sob, nullptr, NROWS, 256, 256, 0);
  gemm(t1, Waw, baw, awb, nullptr, NROWS, 128, 256, 0);
  deform_sample_kernel<<<(BSZ * LQRY * NHEAD) / 4, 128, 0, stream>>>(sob, awb, rp, valh, cab);
  gemm(cab, Wop, bop, capb, nullptr, NROWS, 256, 256, 0);
  add_ln_kernel<<<NROWS, 256, 0, stream>>>(capb, t1, ln2g, ln2b, t2);

  // --- FFN ---
  gemm(t2, W1, bf1, ffh, nullptr, NROWS, DFF, 256, 1);   // fused ReLU
  gemm(ffh, W2, bf2, ffo, nullptr, NROWS, 256, DFF, 0);
  add_ln_kernel<<<NROWS, 256, 0, stream>>>(ffo, t2, ln3g, ln3b, (float*)d_out);
}
